// BitLinear158_51488067944505
// MI455X (gfx1250) — 8-bit matrix operand order repaired and checked by reading, compiled, not run on hardware
//
#include <hip/hip_runtime.h>
#include <stdint.h>

typedef __attribute__((ext_vector_type(8))) int v8i;
typedef __attribute__((ext_vector_type(4))) int v4i;
typedef __attribute__((ext_vector_type(2))) int v2i;

#define IN_F   4096
#define OUT_F  4096
#define MROWS  8192
#define EPS_F  1e-5f
#define NPART  1024


__global__ void k_init(unsigned* wsu) {
    wsu[3] = 0u;
}

__global__ __launch_bounds__(256) void k_wabs(const float* __restrict__ w,
                                              float* __restrict__ partial) {
    __shared__ float sm[256];
    const int tid = threadIdx.x;
    float s = 0.f;
    const int stride = gridDim.x * blockDim.x;
    for (int i = blockIdx.x * blockDim.x + tid; i < IN_F * OUT_F; i += stride)
        s += __builtin_fabsf(w[i]);
    sm[tid] = s;
    __syncthreads();
    for (int o = 128; o > 0; o >>= 1) {
        if (tid < o) sm[tid] += sm[tid + o];
        __syncthreads();
    }
    if (tid == 0) partial[blockIdx.x] = sm[0];
}

__global__ __launch_bounds__(256) void k_xmax(const float* __restrict__ x,
                                              unsigned* __restrict__ maxcell) {
    __shared__ unsigned sm[256];
    const int tid = threadIdx.x;
    float m = 0.f;
    const int stride = gridDim.x * blockDim.x;
    for (int i = blockIdx.x * blockDim.x + tid; i < MROWS * IN_F; i += stride)
        m = fmaxf(m, __builtin_fabsf(x[i]));
    sm[tid] = __float_as_uint(m);
    __syncthreads();
    for (int o = 128; o > 0; o >>= 1) {
        if (tid < o) sm[tid] = sm[tid] > sm[tid + o] ? sm[tid] : sm[tid + o];
        __syncthreads();
    }
    if (tid == 0) atomicMax(maxcell, sm[0]);
}

__global__ __launch_bounds__(256) void k_finalize(float* __restrict__ wsf,
                                                  const float* __restrict__ partial,
                                                  const unsigned* __restrict__ maxcell) {
    __shared__ float sm[256];
    const int tid = threadIdx.x;
    float s = partial[tid] + partial[tid + 256] + partial[tid + 512] + partial[tid + 768];
    sm[tid] = s;
    __syncthreads();
    for (int o = 128; o > 0; o >>= 1) {
        if (tid < o) sm[tid] += sm[tid + o];
        __syncthreads();
    }
    if (tid == 0) {
        float w_scale = fmaxf(sm[0] * (1.0f / (float)(IN_F * OUT_F)), EPS_F);
        float gamma   = fmaxf(__uint_as_float(*maxcell), EPS_F);
        wsf[0] = 1.0f / w_scale;
        wsf[1] = 128.0f / gamma;
        wsf[2] = gamma * w_scale * (1.0f / 128.0f);
    }
}

__device__ __forceinline__ int q_clip(float v, float lo, float hi) {
    return (int)fminf(fmaxf(__builtin_rintf(v), lo), hi);
}

__global__ __launch_bounds__(256) void k_quant_w(const float* __restrict__ w,
                                                  signed char* __restrict__ wq,
                                                  const float* __restrict__ wsf) {
    const float inv = wsf[0];
    const int i = (blockIdx.x * blockDim.x + threadIdx.x) * 4;
    float4 v = *(const float4*)(w + i);
    int a = q_clip(v.x * inv, -1.f, 1.f);
    int b = q_clip(v.y * inv, -1.f, 1.f);
    int c = q_clip(v.z * inv, -1.f, 1.f);
    int d = q_clip(v.w * inv, -1.f, 1.f);
    *(int*)(wq + i) = (a & 0xff) | ((b & 0xff) << 8) | ((c & 0xff) << 16) | (d << 24);
}

__global__ __launch_bounds__(256) void k_quant_x(const float* __restrict__ x,
                                                  signed char* __restrict__ xq,
                                                  const float* __restrict__ wsf) {
    const float sc = wsf[1];
    const int i = (blockIdx.x * blockDim.x + threadIdx.x) * 4;
    float4 v = *(const float4*)(x + i);
    int a = q_clip(v.x * sc, -128.f, 127.f);
    int b = q_clip(v.y * sc, -128.f, 127.f);
    int c = q_clip(v.z * sc, -128.f, 127.f);
    int d = q_clip(v.w * sc, -128.f, 127.f);
    *(int*)(xq + i) = (a & 0xff) | ((b & 0xff) << 8) | ((c & 0xff) << 16) | (d << 24);
}

__global__ __launch_bounds__(256) void k_gemm(const signed char* __restrict__ Xq,
                                              const signed char* __restrict__ Wq,
                                              const float* __restrict__ bias,
                                              const float* __restrict__ wsf,
                                              float* __restrict__ out) {
    const int tid = threadIdx.x;
    const int l   = tid & 31;
    const int wv  = tid >> 5;
    const int h   = l >> 4;
    const int r   = l & 15;
    const int wm  = wv & 3;
    const int wn  = wv >> 2;
    const int Mb  = blockIdx.y * 128 + wm * 32;
    const int Nb  = blockIdx.x * 128 + wn * 64;

    v8i acc[2][4];
#pragma unroll
    for (int mt = 0; mt < 2; ++mt)
#pragma unroll
        for (int nt = 0; nt < 4; ++nt) acc[mt][nt] = (v8i)0;

    const signed char* a0p = Xq + (size_t)(Mb + r) * IN_F + 16 * h;
    const signed char* a1p = a0p + (size_t)16 * IN_F;
    const signed char* bp = Wq + (size_t)(Nb + r) * IN_F + 16 * h;

#pragma unroll 2
    for (int k0 = 0; k0 < IN_F; k0 += 64) {
        v8i a[2], b[4];
        {
            v2i t0 = *(const v2i*)(a0p + k0);
            v2i t1 = *(const v2i*)(a0p + k0 + 8);
            v2i t2 = *(const v2i*)(a0p + k0 + 32);
            v2i t3 = *(const v2i*)(a0p + k0 + 40);
            a[0][0]=t0[0]; a[0][1]=t0[1]; a[0][2]=t1[0]; a[0][3]=t1[1];
            a[0][4]=t2[0]; a[0][5]=t2[1]; a[0][6]=t3[0]; a[0][7]=t3[1];
        }
        {
            v2i t0 = *(const v2i*)(a1p + k0);
            v2i t1 = *(const v2i*)(a1p + k0 + 8);
            v2i t2 = *(const v2i*)(a1p + k0 + 32);
            v2i t3 = *(const v2i*)(a1p + k0 + 40);
            a[1][0]=t0[0]; a[1][1]=t0[1]; a[1][2]=t1[0]; a[1][3]=t1[1];
            a[1][4]=t2[0]; a[1][5]=t2[1]; a[1][6]=t3[0]; a[1][7]=t3[1];
        }
#pragma unroll
        for (int nt = 0; nt < 4; ++nt) {
            const signed char* p = bp + (size_t)(16 * nt) * IN_F + k0;
            v4i u0 = *(const v4i*)(p);
            v4i u1 = *(const v4i*)(p + 32);
            b[nt][0]=u0[0]; b[nt][1]=u0[1]; b[nt][2]=u0[2]; b[nt][3]=u0[3];
            b[nt][4]=u1[0]; b[nt][5]=u1[1]; b[nt][6]=u1[2]; b[nt][7]=u1[3];
        }
#pragma unroll
        for (int nt = 0; nt < 4; ++nt) {
            acc[0][nt] = __builtin_amdgcn_wmma_i32_16x16x64_iu8(true, a[0], true, b[nt],
                                                                acc[0][nt], false, false);
            acc[1][nt] = __builtin_amdgcn_wmma_i32_16x16x64_iu8(true, a[1], true, b[nt],
                                                                acc[1][nt], false, false);
        }
    }

    const float scale = wsf[2];
#pragma unroll
    for (int nt = 0; nt < 4; ++nt) {
        const int   n  = Nb + 16 * nt + r;
        const float bv = bias[n];
#pragma unroll
        for (int mt = 0; mt < 2; ++mt) {
            float* o = out + (size_t)(Mb + 16 * mt + 8 * h) * OUT_F + n;
            v8i c = acc[mt][nt];
#pragma unroll
            for (int rr = 0; rr < 8; ++rr)
                o[(size_t)rr * OUT_F] = (float)c[rr] * scale + bv;
        }
    }
}

extern "C" void kernel_launch(void* const* d_in, const int* in_sizes, int n_in,
                              void* d_out, int out_size, void* d_ws, size_t ws_size,
                              hipStream_t stream) {
    const float* x    = (const float*)d_in[0];
    const float* w    = (const float*)d_in[1];
    const float* bias = (const float*)d_in[2];
    float* out = (float*)d_out;

    float*    wsf     = (float*)d_ws;
    unsigned* wsu     = (unsigned*)d_ws;
    float*    partial = (float*)((char*)d_ws + 64);
    signed char* Wq   = (signed char*)d_ws + 8192;
    signed char* Xq   = Wq + (size_t)IN_F * OUT_F;

    k_init<<<1, 1, 0, stream>>>(wsu);
    k_wabs<<<NPART, 256, 0, stream>>>(w, partial);
    k_xmax<<<2048, 256, 0, stream>>>(x, wsu + 3);
    k_finalize<<<1, 256, 0, stream>>>(wsf, partial, wsu + 3);
    k_quant_w<<<(IN_F * OUT_F) / (256 * 4), 256, 0, stream>>>(w, Wq, wsf);
    k_quant_x<<<(MROWS * IN_F) / (256 * 4), 256, 0, stream>>>(x, Xq, wsf);

    dim3 grid(OUT_F / 128, MROWS / 128);
    k_gemm<<<grid, 256, 0, stream>>>(Xq, Wq, bias, wsf, out);
}
